// LayerConnAttention_56066503082542
// MI455X (gfx1250) — matrix operand order repaired and checked by reading, not compiled, not run on hardware
//
#include <hip/hip_runtime.h>
#include <hip/hip_bf16.h>

// ---------------------------------------------------------------------------
// CDNA5 (gfx1250) wave32 WMMA building blocks — bf16 end-to-end pipeline
// ---------------------------------------------------------------------------
typedef __attribute__((ext_vector_type(16))) __bf16 v16bf;
typedef __attribute__((ext_vector_type(8)))  float  v8f;
typedef __attribute__((ext_vector_type(4)))  int    v4i;

union BF16Frag { v16bf v; unsigned short u[16]; uint4 q[2]; };

__device__ __forceinline__ unsigned short f2bf(float f) {
    union { float f; unsigned int u; } x; x.f = f;
    unsigned int r = x.u + 0x7FFFu + ((x.u >> 16) & 1u);   // RNE
    return (unsigned short)(r >> 16);
}

__device__ __forceinline__ v8f wmma_bf16(v16bf a, v16bf b, v8f c) {
    return __builtin_amdgcn_wmma_f32_16x16x32_bf16(
        false, a, false, b, (short)0, c, false, false);
}

// --- gfx1250 async global->LDS copy (probe-confirmed working) --------------
#if __has_builtin(__builtin_amdgcn_global_load_async_to_lds_b128)
#define HAVE_ASYNC 1
typedef __attribute__((address_space(1))) v4i* as1_v4i;   // global int4*
typedef __attribute__((address_space(3))) v4i* as3_v4i;   // LDS int4*
#else
#define HAVE_ASYNC 0
#endif

__device__ __forceinline__ void cp16(void* lds_dst, const void* gsrc) {
#if HAVE_ASYNC
    __builtin_amdgcn_global_load_async_to_lds_b128(
        (as1_v4i)(v4i*)gsrc, (as3_v4i)(v4i*)lds_dst, 0, 0);
#else
    *(uint4*)lds_dst = *(const uint4*)gsrc;
#endif
}

// Wait until at most N async-to-LDS ops remain in flight (in-order complete).
template <int N>
__device__ __forceinline__ void async_wait() {
#if HAVE_ASYNC
#if __has_builtin(__builtin_amdgcn_s_wait_asynccnt)
    __builtin_amdgcn_s_wait_asynccnt(N);
#else
    asm volatile("s_wait_asynccnt %0" :: "i"(N) : "memory");
#endif
#endif
}

// Problem constants
#define NB      4
#define LQ      1024
#define LK      1024
#define NHEAD   8
#define DK      64
#define DMODEL  512
#define MROWS   (NB * LQ)          // 4096
#define TOPK    6
#define EPS_F   1e-7f

// ---------------------------------------------------------------------------
// f32 -> bf16 elementwise (bandwidth bound; n % 4 == 0 for all our tensors)
// ---------------------------------------------------------------------------
__global__ __launch_bounds__(256)
void cvt_f32_bf16(const float* __restrict__ src, unsigned short* __restrict__ dst,
                  int n)
{
    const int i = (blockIdx.x * 256 + threadIdx.x) * 4;
    if (i < n) {
        const float4 v = *(const float4*)(src + i);
        ushort4 o;
        o.x = f2bf(v.x); o.y = f2bf(v.y); o.z = f2bf(v.z); o.w = f2bf(v.w);
        *(ushort4*)(dst + i) = o;
    }
}

// ---------------------------------------------------------------------------
// GEMM: Y[M,N] = X[M,K] @ W[N,K]^T + bias[N]   (bf16 in, bf16 or f32 out)
// 64x64 block tile, BK=32, 256 threads = 8 waves; wave -> 16x32 of C.
// Ping-pong LDS: async-copy tile k+1 overlaps WMMA on tile k
// (s_wait_asynccnt 2 = current tile landed, next still in flight).
// ---------------------------------------------------------------------------
#define GBM 64
#define GBN 64
#define GBK 32

__global__ __launch_bounds__(256)
void gemm_bf16(const unsigned short* __restrict__ X,
               const unsigned short* __restrict__ W,
               const float* __restrict__ bias,
               float* __restrict__ Yf, unsigned short* __restrict__ Yb,
               int M, int N, int K)
{
    __shared__ __align__(16) unsigned short Xs[2][GBM][GBK];   // 8 KB
    __shared__ __align__(16) unsigned short Ws[2][GBN][GBK];   // 8 KB

    const int tid  = threadIdx.x;
    const int lane = tid & 31;
    const int wave = tid >> 5;
    const int lm   = lane & 15;
    const int lh   = lane >> 4;
    const int bm   = blockIdx.y * GBM;
    const int bn   = blockIdx.x * GBN;
    const int mi   = wave & 3;        // M sub-tile (16 rows)
    const int nj   = wave >> 2;       // N sub-pair (32 cols)

    const int sr = tid >> 2;          // staging row 0..63
    const int sc = (tid & 3) * 8;     // staging col chunk (8 bf16 = 16 B)

    const unsigned short* Xrow = X + (size_t)(bm + sr) * K + sc;
    const unsigned short* Wrow = W + (size_t)(bn + sr) * K + sc;

    v8f acc0 = {}; v8f acc1 = {};

    const int NT = K / GBK;
    // prologue: stage tile 0 into buffer 0
    cp16(&Xs[0][sr][sc], Xrow);
    cp16(&Ws[0][sr][sc], Wrow);

    for (int kt = 0; kt < NT; ++kt) {
        const int cur = kt & 1;
        if (kt + 1 < NT) {
            const int nxt = cur ^ 1;
            cp16(&Xs[nxt][sr][sc], Xrow + (kt + 1) * GBK);
            cp16(&Ws[nxt][sr][sc], Wrow + (kt + 1) * GBK);
            if (kt + 2 < NT) {   // global_prefetch_b8 two tiles ahead
                __builtin_prefetch(Xrow + (kt + 2) * GBK, 0, 1);
                __builtin_prefetch(Wrow + (kt + 2) * GBK, 0, 1);
            }
            async_wait<2>();     // tile kt landed; tile kt+1 in flight
        } else {
            async_wait<0>();
        }
        __syncthreads();

        BF16Frag a, b0, b1;
        // A: lane = m + 16*(k>=16); e = k%16
        a.q[0] = *(const uint4*)&Xs[cur][mi * 16 + lm][lh * 16];
        a.q[1] = *(const uint4*)&Xs[cur][mi * 16 + lm][lh * 16 + 8];
        // B: lane = n + 16*(k>=16); e = k%16
        b0.q[0] = *(const uint4*)&Ws[cur][nj * 32 + lm][lh * 16];
        b0.q[1] = *(const uint4*)&Ws[cur][nj * 32 + lm][lh * 16 + 8];
        b1.q[0] = *(const uint4*)&Ws[cur][nj * 32 + lm + 16][lh * 16];
        b1.q[1] = *(const uint4*)&Ws[cur][nj * 32 + lm + 16][lh * 16 + 8];

        acc0 = wmma_bf16(a.v, b0.v, acc0);
        acc1 = wmma_bf16(a.v, b1.v, acc1);
        __syncthreads();   // protects buffer cur from overwrite at kt+1
    }

    // C/D layout: VGPR r -> row (r + 8*lh), col = lane%16
    const int mrow  = bm + mi * 16 + lh * 8;
    const int ncol0 = bn + nj * 32 + lm;
    const float bia0 = bias[ncol0];
    const float bia1 = bias[ncol0 + 16];
    #pragma unroll
    for (int r = 0; r < 8; ++r) {
        const int m  = mrow + r;
        const float y0 = acc0[r] + bia0;
        const float y1 = acc1[r] + bia1;
        if (Yb) {
            Yb[(size_t)m * N + ncol0]      = f2bf(y0);
            Yb[(size_t)m * N + ncol0 + 16] = f2bf(y1);
        } else {
            Yf[(size_t)m * N + ncol0]      = y0;
            Yf[(size_t)m * N + ncol0 + 16] = y1;
        }
    }
}

// ---------------------------------------------------------------------------
// Attention: block = one (h*B+b, 16-row q tile).
//   Phase 1: S = (Q Kt)/8 via WMMA -> LDS f32
//   Phase 2: softmax + top-6 threshold sparsify + renorm; write attn (f32),
//            stash attn as bf16 in LDS for phase 3
//   Phase 3: ctx = attn @ V via WMMA; V tiles double-buffered via async-to-LDS
// ---------------------------------------------------------------------------
#define QT 16

__global__ __launch_bounds__(256)
void attn_kernel(const unsigned short* __restrict__ Qb,
                 const unsigned short* __restrict__ Kb,
                 const unsigned short* __restrict__ Vb,
                 float* __restrict__ attn_out,
                 unsigned short* __restrict__ ctxb)
{
    extern __shared__ char smemraw[];
    float* S        = (float*)smemraw;       // QT*1024 f32   (64 KB)
    float* red      = S + QT * LK;           // QT*16
    float* top6buf  = red + QT * 16;         // QT*96
    float* partial  = top6buf + QT * 96;     // 4*256
    float* rowmax   = partial + 4 * 256;     // QT
    float* rowsumA  = rowmax + QT;           // QT
    float* rowdelta = rowsumA + QT;          // QT
    float* rowwsum  = rowdelta + QT;         // QT
    unsigned short* Abf = (unsigned short*)(rowwsum + QT);  // QT*1024 bf16 (32 KB)
    unsigned short* Vs0 = Abf + QT * LK;                    // 64*64 bf16   (8 KB)
    unsigned short* Vs1 = Vs0 + 64 * 64;                    // 64*64 bf16   (8 KB)

    const int tid  = threadIdx.x;
    const int lane = tid & 31;
    const int wave = tid >> 5;
    const int lm   = lane & 15;
    const int lh   = lane >> 4;

    const int bh = blockIdx.y;               // = h*B + b
    const int h  = bh >> 2;
    const int b  = bh & 3;
    const int q0 = blockIdx.x * QT;
    const float inv_temp = 0.125f;           // 1/sqrt(64)

    // ---- Phase 1: scores -------------------------------------------------
    BF16Frag aq[2];
    {
        const unsigned short* qrow =
            Qb + ((size_t)(b * LQ + q0 + lm)) * DMODEL + h * DK;
        #pragma unroll
        for (int kc = 0; kc < 2; ++kc) {
            aq[kc].q[0] = *(const uint4*)(qrow + kc * 32 + lh * 16);
            aq[kc].q[1] = *(const uint4*)(qrow + kc * 32 + lh * 16 + 8);
        }
    }
    #pragma unroll
    for (int t = 0; t < 8; ++t) {
        const int jt = wave + 8 * t;         // 0..63 (16 lk cols each)
        const unsigned short* krow =
            Kb + ((size_t)(b * LK + jt * 16 + lm)) * DMODEL + h * DK;
        v8f acc = {};
        #pragma unroll
        for (int kc = 0; kc < 2; ++kc) {
            BF16Frag bk;
            bk.q[0] = *(const uint4*)(krow + kc * 32 + lh * 16);
            bk.q[1] = *(const uint4*)(krow + kc * 32 + lh * 16 + 8);
            acc = wmma_bf16(aq[kc].v, bk.v, acc);
        }
        #pragma unroll
        for (int r = 0; r < 8; ++r)
            S[(r + 8 * lh) * LK + jt * 16 + lm] = acc[r] * inv_temp;
    }
    __syncthreads();

    // ---- Phase 2: softmax + top-6 sparsify -------------------------------
    const int row = tid >> 4;                // 16 threads per row
    const int sub = tid & 15;
    float* Srow = S + row * LK;

    float mx = -3.4e38f;
    for (int i = 0; i < 64; ++i) mx = fmaxf(mx, Srow[sub + (i << 4)]);
    red[row * 16 + sub] = mx;
    __syncthreads();
    if (sub == 0) {
        float m2 = red[row * 16];
        for (int i = 1; i < 16; ++i) m2 = fmaxf(m2, red[row * 16 + i]);
        rowmax[row] = m2;
    }
    __syncthreads();
    mx = rowmax[row];

    float sum = 0.f;
    for (int i = 0; i < 64; ++i) {
        const int c = sub + (i << 4);
        const float p = __expf(Srow[c] - mx);
        Srow[c] = p;
        sum += p;
    }
    red[row * 16 + sub] = sum;
    __syncthreads();
    if (sub == 0) {
        float s2 = 0.f;
        for (int i = 0; i < 16; ++i) s2 += red[row * 16 + i];
        rowsumA[row] = s2;
    }
    __syncthreads();
    const float invsum = 1.0f / rowsumA[row];

    float t6[TOPK] = {-1.f, -1.f, -1.f, -1.f, -1.f, -1.f};
    for (int i = 0; i < 64; ++i) {
        const float p = Srow[sub + (i << 4)];
        if (p > t6[TOPK - 1]) {
            int j = TOPK - 1;
            while (j > 0 && p > t6[j - 1]) { t6[j] = t6[j - 1]; --j; }
            t6[j] = p;
        }
    }
    #pragma unroll
    for (int i = 0; i < TOPK; ++i) top6buf[row * 96 + sub * TOPK + i] = t6[i];
    __syncthreads();
    if (sub == 0) {
        float best[TOPK] = {-1.f, -1.f, -1.f, -1.f, -1.f, -1.f};
        for (int i = 0; i < 96; ++i) {
            const float p = top6buf[row * 96 + i];
            if (p > best[TOPK - 1]) {
                int j = TOPK - 1;
                while (j > 0 && p > best[j - 1]) { best[j] = best[j - 1]; --j; }
                best[j] = p;
            }
        }
        rowdelta[row] = best[TOPK - 1] * invsum + EPS_F;
    }
    __syncthreads();
    const float delta = rowdelta[row];

    float wsum = 0.f;
    for (int i = 0; i < 64; ++i) {
        const int c = sub + (i << 4);
        const float w = fmaxf(Srow[c] * invsum - delta, 0.f);
        Srow[c] = w;
        wsum += w;
    }
    red[row * 16 + sub] = wsum;
    __syncthreads();
    if (sub == 0) {
        float s2 = 0.f;
        for (int i = 0; i < 16; ++i) s2 += red[row * 16 + i];
        rowwsum[row] = s2;
    }
    __syncthreads();
    const float wnorm = 1.0f / (rowwsum[row] + EPS_F);
    float* arow = attn_out + ((size_t)bh * LQ + (q0 + row)) * LK;
    for (int i = 0; i < 64; ++i) {
        const int c  = sub + (i << 4);
        const float av = Srow[c] * wnorm;
        arow[c] = av;                        // coalesced f32 output
        Abf[row * LK + c] = f2bf(av);        // bf16 copy for phase 3
    }
    __syncthreads();

    // ---- Phase 3: ctx = attn(16x1024) @ V(1024x64) -----------------------
    const int ntile = wave & 3;              // 16-col tile of d_v
    const int khalf = wave >> 2;             // K split halves
    const int ncol  = ntile * 16 + lm;

    // stage V rows {ks*32..+31} and {512+ks*32..+31} (both K-halves) to dst
    auto stageV = [&](int ks, unsigned short* dst) {
        int i = tid;
        #pragma unroll
        for (int it = 0; it < 2; ++it, i += 256) {
            const int r = i >> 3, c = (i & 7) * 8;
            const int krow = (r < 32) ? (ks * 32 + r) : (512 + ks * 32 + (r - 32));
            cp16(dst + r * 64 + c,
                 Vb + ((size_t)(b * LK + krow)) * DMODEL + h * DK + c);
        }
    };

    v8f oacc = {};
    stageV(0, Vs0);                          // prologue
    for (int ks = 0; ks < 16; ++ks) {
        unsigned short* VsC = (ks & 1) ? Vs1 : Vs0;
        if (ks + 1 < 16) {
            stageV(ks + 1, (ks & 1) ? Vs0 : Vs1);
            async_wait<2>();                 // tile ks landed, ks+1 in flight
        } else {
            async_wait<0>();
        }
        __syncthreads();

        const int kbA = khalf * 512 + ks * 32;
        BF16Frag aa, bv;
        aa.q[0] = *(const uint4*)&Abf[lm * LK + kbA + lh * 16];
        aa.q[1] = *(const uint4*)&Abf[lm * LK + kbA + lh * 16 + 8];
        #pragma unroll
        for (int e = 0; e < 16; ++e)
            bv.u[e] = VsC[(khalf * 32 + lh * 16 + e) * 64 + ncol];
        oacc = wmma_bf16(aa.v, bv.v, oacc);
        __syncthreads();                     // protect VsC from next overwrite
    }
    if (wave >= 4) {                         // stash upper-K partials
        float* p = partial + (wave - 4) * 256;
        #pragma unroll
        for (int r = 0; r < 8; ++r) p[r * 32 + lane] = oacc[r];
    }
    __syncthreads();
    if (wave < 4) {                          // combine halves, write bf16 ctx
        const float* p = partial + wave * 256;
        #pragma unroll
        for (int r = 0; r < 8; ++r) {
            const float val = oacc[r] + p[r * 32 + lane];
            const int m = q0 + r + 8 * lh;
            ctxb[((size_t)(b * LQ + m)) * DMODEL + h * DK + ncol] = f2bf(val);
        }
    }
}

// ---------------------------------------------------------------------------
// output = sigmoid(gate_lin) * tanh(fc_out)
// ---------------------------------------------------------------------------
__global__ __launch_bounds__(256)
void gate_tanh_kernel(const float* __restrict__ F, const float* __restrict__ G,
                      float* __restrict__ out, int n)
{
    const int i = blockIdx.x * 256 + threadIdx.x;
    if (i < n) {
        const float g = 1.0f / (1.0f + __expf(-G[i]));
        out[i] = g * tanhf(F[i]);
    }
}

// ---------------------------------------------------------------------------
extern "C" void kernel_launch(void* const* d_in, const int* in_sizes, int n_in,
                              void* d_out, int out_size, void* d_ws, size_t ws_size,
                              hipStream_t stream)
{
    const float* q   = (const float*)d_in[0];
    const float* k   = (const float*)d_in[1];
    const float* v   = (const float*)d_in[2];
    const float* Wq  = (const float*)d_in[3];
    const float* bq  = (const float*)d_in[4];
    const float* Wk  = (const float*)d_in[5];
    const float* bk  = (const float*)d_in[6];
    const float* Wv  = (const float*)d_in[7];
    const float* bv  = (const float*)d_in[8];
    const float* Wfc = (const float*)d_in[9];
    const float* bfc = (const float*)d_in[10];
    const float* Wg  = (const float*)d_in[11];
    const float* bg  = (const float*)d_in[12];

    float* out  = (float*)d_out;                       // (4,1024,512)
    float* attn = out + (size_t)NB * LQ * DMODEL;      // (32,1024,1024)

    // ---- workspace carve (bytes; every size is 16B-aligned) --------------
    char* wsb = (char*)d_ws;
    const size_t XSZ = (size_t)MROWS * DMODEL;         // 4096*512
    const size_t WSZ = (size_t)DMODEL * DMODEL;        // 512*512
    unsigned short* qb   = (unsigned short*)wsb; wsb += XSZ * 2;
    unsigned short* kb   = (unsigned short*)wsb; wsb += XSZ * 2;
    unsigned short* vb   = (unsigned short*)wsb; wsb += XSZ * 2;
    unsigned short* wqb  = (unsigned short*)wsb; wsb += WSZ * 2;
    unsigned short* wkb  = (unsigned short*)wsb; wsb += WSZ * 2;
    unsigned short* wvb  = (unsigned short*)wsb; wsb += WSZ * 2;
    unsigned short* wfcb = (unsigned short*)wsb; wsb += WSZ * 2;
    unsigned short* wgb  = (unsigned short*)wsb; wsb += WSZ * 2;
    unsigned short* Qb   = (unsigned short*)wsb; wsb += XSZ * 2;
    unsigned short* Kb   = (unsigned short*)wsb; wsb += XSZ * 2;
    unsigned short* Vb   = (unsigned short*)wsb; wsb += XSZ * 2;
    unsigned short* ctxb = (unsigned short*)wsb; wsb += XSZ * 2;
    float*          Ff   = (float*)wsb;          wsb += XSZ * 4;
    float*          Gg   = (float*)wsb;          wsb += XSZ * 4;

    const dim3 blk(256);

    // ---- 1) f32 -> bf16 conversions --------------------------------------
    const int nx = (int)XSZ, nw = (int)WSZ;
    cvt_f32_bf16<<<nx / 4 / 256, blk, 0, stream>>>(q,   qb,   nx);
    cvt_f32_bf16<<<nx / 4 / 256, blk, 0, stream>>>(k,   kb,   nx);
    cvt_f32_bf16<<<nx / 4 / 256, blk, 0, stream>>>(v,   vb,   nx);
    cvt_f32_bf16<<<nw / 4 / 256, blk, 0, stream>>>(Wq,  wqb,  nw);
    cvt_f32_bf16<<<nw / 4 / 256, blk, 0, stream>>>(Wk,  wkb,  nw);
    cvt_f32_bf16<<<nw / 4 / 256, blk, 0, stream>>>(Wv,  wvb,  nw);
    cvt_f32_bf16<<<nw / 4 / 256, blk, 0, stream>>>(Wfc, wfcb, nw);
    cvt_f32_bf16<<<nw / 4 / 256, blk, 0, stream>>>(Wg,  wgb,  nw);

    // ---- 2) QKV projections (bf16 out) -----------------------------------
    const dim3 ggrid(DMODEL / GBN, MROWS / GBM);       // (8, 64)
    gemm_bf16<<<ggrid, blk, 0, stream>>>(qb, wqb, bq, nullptr, Qb,
                                         MROWS, DMODEL, DMODEL);
    gemm_bf16<<<ggrid, blk, 0, stream>>>(kb, wkb, bk, nullptr, Kb,
                                         MROWS, DMODEL, DMODEL);
    gemm_bf16<<<ggrid, blk, 0, stream>>>(vb, wvb, bv, nullptr, Vb,
                                         MROWS, DMODEL, DMODEL);

    // ---- 3) attention + top-6 sparsify + attn@V --------------------------
    const size_t smem_bytes =
        (QT * LK + QT * 16 + QT * 96 + 4 * 256 + 4 * QT) * sizeof(float)  // f32 area
        + QT * LK * 2                                                      // Abf
        + 2 * 64 * 64 * 2;                                                 // Vs0/Vs1
    attn_kernel<<<dim3(LQ / QT, NHEAD * NB), blk, smem_bytes, stream>>>(
        Qb, Kb, Vb, attn, ctxb);

    // ---- 4) fc + gate GEMMs (f32 out) ------------------------------------
    gemm_bf16<<<ggrid, blk, 0, stream>>>(ctxb, wfcb, bfc, Ff, nullptr,
                                         MROWS, DMODEL, DMODEL);
    gemm_bf16<<<ggrid, blk, 0, stream>>>(ctxb, wgb,  bg,  Gg, nullptr,
                                         MROWS, DMODEL, DMODEL);

    // ---- 5) elementwise gate ---------------------------------------------
    const int ne = (int)(NB * LQ * DMODEL);
    gate_tanh_kernel<<<(ne + 255) / 256, blk, 0, stream>>>(Ff, Gg, out, ne);
}
